// WindowAttention_16097537425688
// MI455X (gfx1250) — matrix operand order repaired and checked by reading, not compiled, not run on hardware
//
#include <hip/hip_runtime.h>
#include <hip/hip_bf16.h>

// CDNA5 / gfx1250 wave32 WMMA window-attention.
// One workgroup (256 threads = 8 wave32) per window. All intermediates in LDS.

typedef __attribute__((ext_vector_type(16))) _Float16 v16h;
typedef __attribute__((ext_vector_type(8)))  _Float16 v8h;
typedef __attribute__((ext_vector_type(4)))  _Float16 v4h;
typedef __attribute__((ext_vector_type(8)))  float    v8f;
typedef __attribute__((ext_vector_type(4)))  float    v4f;

#define NTOK  64     // tokens per window
#define CDIM  96     // embedding dim
#define NHEAD 3
#define HDIM  32

#define QKV_W_ELEMS  (3*CDIM*CDIM)     // 27648
#define PROJ_W_ELEMS (CDIM*CDIM)       // 9216
#define BIAS3_ELEMS  (NHEAD*NTOK*NTOK) // 12288

// ---- WMMA fragment loaders (ISA 7.12.2 layouts, 16-bit A 16x32 / B 32x16) ----

// A fragment: A[m][k], source row-major with row stride ld (halfs).
// lanes 0-15: row m = mbase+lane, K chunks {kb..kb+7} -> V0-3, {kb+16..kb+23} -> V4-7
// lanes 16-31: same rows, K chunks shifted by +8.
__device__ __forceinline__ v16h frag_a(const _Float16* base, int ld,
                                       int mbase, int kbase, int lane) {
  const int hh = lane >> 4;
  const int m  = lane & 15;
  const _Float16* p = base + (mbase + m) * ld + kbase + hh * 8;
  union { v16h v; v8h h[2]; } u;
  u.h[0] = *(const v8h*)(p);        // K = kb + hh*8 .. +7
  u.h[1] = *(const v8h*)(p + 16);   // K = kb + 16 + hh*8 .. +7
  return u.v;
}

// B fragment: B[k][n] read from a row-major source S with S[n][k] (i.e. B = S^T).
// lanes 0-15 hold K=0..7,16..23; lanes 16-31 hold K=8..15,24..31 (matches frag_a).
__device__ __forceinline__ v16h frag_b(const _Float16* base, int ld,
                                       int nbase, int kbase, int lane) {
  const int hh = lane >> 4;
  const int n  = lane & 15;
  const _Float16* p = base + (nbase + n) * ld + kbase + hh * 8;
  union { v16h v; v8h h[2]; } u;
  u.h[0] = *(const v8h*)(p);
  u.h[1] = *(const v8h*)(p + 16);
  return u.v;
}

__device__ __forceinline__ v8f wmma16(v16h a, v16h b, v8f c) {
  return __builtin_amdgcn_wmma_f32_16x16x32_f16(false, a, false, b,
                                                (short)0, c, false, false);
}

__device__ __forceinline__ v8f splat8(float v) {
  v8f r;
  #pragma unroll
  for (int i = 0; i < 8; ++i) r[i] = v;
  return r;
}

// ---- prep: fp32 weights -> f16, gather relative-position bias to [3][64][64] ----
__global__ void winattn_prep_kernel(const float* __restrict__ qkv_w,
                                    const float* __restrict__ proj_w,
                                    const float* __restrict__ bias_table,
                                    const int*   __restrict__ rel_idx,
                                    _Float16* __restrict__ qkvw_h,
                                    _Float16* __restrict__ projw_h,
                                    float* __restrict__ bias3) {
  int idx = blockIdx.x * blockDim.x + threadIdx.x;
  if (idx < QKV_W_ELEMS) {
    qkvw_h[idx] = (_Float16)qkv_w[idx];
  } else if (idx < QKV_W_ELEMS + PROJ_W_ELEMS) {
    int j = idx - QKV_W_ELEMS;
    projw_h[j] = (_Float16)proj_w[j];
  } else if (idx < QKV_W_ELEMS + PROJ_W_ELEMS + BIAS3_ELEMS) {
    int j  = idx - (QKV_W_ELEMS + PROJ_W_ELEMS);
    int h  = j / (NTOK*NTOK);
    int qk = j % (NTOK*NTOK);
    bias3[h*(NTOK*NTOK) + qk] = bias_table[rel_idx[qk]*NHEAD + h];
  }
}

// ---- main: one workgroup per window ----
__global__ void __launch_bounds__(256)
winattn_kernel(const float* __restrict__ x, const float* __restrict__ mask,
               const float* __restrict__ qkv_b, const float* __restrict__ proj_b,
               const _Float16* __restrict__ qkvw, const _Float16* __restrict__ projw,
               const float* __restrict__ bias3, float* __restrict__ out, int nmask) {
  extern __shared__ char smem[];
  _Float16* Xh = (_Float16*)smem;                 // [64][96] f16
  _Float16* Qh = Xh + NTOK*CDIM;                  // [64][96] f16 (pre-scaled)
  _Float16* Kh = Qh + NTOK*CDIM;                  // [64][96] f16
  _Float16* Vt = Kh + NTOK*CDIM;                  // [96][64] f16 (transposed V)
  float*    S  = (float*)(Vt + NTOK*CDIM);        // [3][64][64] f32 logits
  _Float16* P  = (_Float16*)(S + NHEAD*NTOK*NTOK);// [3][64][64] f16 probs
  _Float16* Oh = P + NHEAD*NTOK*NTOK;             // [64][96] f16 attn output

  const int w    = blockIdx.x;
  const int tid  = threadIdx.x;
  const int lane = tid & 31;
  const int wv   = tid >> 5;
  const int wm   = w & (nmask - 1);               // nmask is a power of two (4096)
  const float scale = 0.17677669529663687f;       // 1/sqrt(32)

  // ---- Phase 1: X -> LDS (f16), vectorized float4 -> half4 ----
  const v4f* xg4 = (const v4f*)(x + (size_t)w * (NTOK*CDIM));
  v4h* Xh4 = (v4h*)Xh;
  #pragma unroll
  for (int i = 0; i < (NTOK*CDIM)/(256*4); ++i) {
    const int idx = tid + i * 256;
    v4f t = xg4[idx];
    v4h h;
    #pragma unroll
    for (int j = 0; j < 4; ++j) h[j] = (_Float16)t[j];
    Xh4[idx] = h;
  }
  __syncthreads();

  // ---- Phase 2: QKV = X @ Wqkv^T + b ; 4 M-tiles x 18 O-tiles = 72 tiles ----
  for (int ti = wv; ti < 72; ti += 8) {
    const int ot = __builtin_amdgcn_readfirstlane(ti / 4);   // wave-uniform
    const int mt = ti % 4;
    const int mb = mt * 16, ob = ot * 16;
    v8f c = splat8(qkv_b[ob + (lane & 15)]);
    #pragma unroll
    for (int kb = 0; kb < CDIM; kb += 32) {
      v16h a = frag_a(Xh,   CDIM, mb, kb, lane);
      v16h b = frag_b(qkvw, CDIM, ob, kb, lane);   // B[k][o] = Wqkv[o][k]
      c = wmma16(a, b, c);
    }
    const int oc = ob + (lane & 15);
    const int m0 = mb + ((lane >> 4) << 3);
    if (ot < 6) {            // Q: store (x@W + b) * scale, row-major [t][96]
      #pragma unroll
      for (int r = 0; r < 8; ++r)
        Qh[(m0 + r) * CDIM + oc] = (_Float16)(c[r] * scale);
    } else if (ot < 12) {    // K: row-major [t][96]
      #pragma unroll
      for (int r = 0; r < 8; ++r)
        Kh[(m0 + r) * CDIM + (oc - CDIM)] = (_Float16)c[r];
    } else {                 // V: transposed [d][t] (column-contiguous -> b128)
      #pragma unroll
      for (int r = 0; r < 8; ++r)
        Vt[(oc - 2*CDIM) * NTOK + (m0 + r)] = (_Float16)c[r];
    }
  }
  __syncthreads();

  // ---- Phase 3: S_h = Q_h @ K_h^T ; 3 heads x 4x4 tiles = 48 tiles ----
  for (int ti = wv; ti < 48; ti += 8) {
    const int h  = __builtin_amdgcn_readfirstlane(ti >> 4);
    const int tt = ti & 15;
    const int mb = (tt >> 2) * 16, nb = (tt & 3) * 16;
    v16h a = frag_a(Qh + h*HDIM, CDIM, mb, 0, lane);
    v16h b = frag_b(Kh + h*HDIM, CDIM, nb, 0, lane); // B[d][kt] = K[kt][d]
    v8f  c = wmma16(a, b, splat8(0.0f));
    const int kk = nb + (lane & 15);
    const int q0 = mb + ((lane >> 4) << 3);
    #pragma unroll
    for (int r = 0; r < 8; ++r)
      S[h*(NTOK*NTOK) + (q0 + r)*NTOK + kk] = c[r];
  }
  __syncthreads();

  // ---- Softmax rows (+ rel-pos bias + shift mask): 192 rows, one per thread ----
  // Rows are 256B-aligned: use float4 for LDS/global traffic (b128s).
  if (tid < NHEAD * NTOK) {
    const int h = tid / NTOK, q = tid % NTOK;
    v4f*       srow = (v4f*)(S     + h*(NTOK*NTOK) + q*NTOK);
    const v4f* brow = (const v4f*)(bias3 + h*(NTOK*NTOK) + q*NTOK);
    const v4f* mrow = (const v4f*)(mask  + (size_t)wm*(NTOK*NTOK) + q*NTOK);
    float mx = -3.0e38f;
    #pragma unroll
    for (int i = 0; i < NTOK/4; ++i) {
      v4f v = srow[i] + brow[i] + mrow[i];
      srow[i] = v;
      mx = fmaxf(mx, fmaxf(fmaxf(v[0], v[1]), fmaxf(v[2], v[3])));
    }
    float sum = 0.0f;
    #pragma unroll
    for (int i = 0; i < NTOK/4; ++i) {
      v4f v = srow[i];
      v4f e;
      #pragma unroll
      for (int j = 0; j < 4; ++j) e[j] = __expf(v[j] - mx);
      srow[i] = e;
      sum += e[0] + e[1] + e[2] + e[3];
    }
    const float inv = 1.0f / sum;
    v4h* prow = (v4h*)(P + h*(NTOK*NTOK) + q*NTOK);
    #pragma unroll
    for (int i = 0; i < NTOK/4; ++i) {
      v4f v = srow[i];
      v4h ph;
      #pragma unroll
      for (int j = 0; j < 4; ++j) ph[j] = (_Float16)(v[j] * inv);
      prow[i] = ph;
    }
  }
  __syncthreads();

  // ---- Phase 4: O_h = P_h @ V_h ; 3 heads x 4x2 tiles = 24 tiles ----
  for (int ti = wv; ti < 24; ti += 8) {
    const int h  = __builtin_amdgcn_readfirstlane(ti / 8);
    const int tt = ti % 8;
    const int mb = (tt >> 1) * 16, nb = (tt & 1) * 16;
    v8f c = splat8(0.0f);
    #pragma unroll
    for (int kb = 0; kb < NTOK; kb += 32) {
      v16h a = frag_a(P + h*(NTOK*NTOK), NTOK, mb, kb, lane);
      v16h b = frag_b(Vt + h*HDIM*NTOK,  NTOK, nb, kb, lane); // B[kt][d] = Vt[d][kt]
      c = wmma16(a, b, c);
    }
    const int d  = nb + (lane & 15);
    const int t0 = mb + ((lane >> 4) << 3);
    #pragma unroll
    for (int r = 0; r < 8; ++r)
      Oh[(t0 + r) * CDIM + h*HDIM + d] = (_Float16)c[r];
  }
  __syncthreads();

  // ---- Phase 5: Out = Oh @ Wproj^T + b ; 4x6 = 24 tiles ----
  float* og = out + (size_t)w * (NTOK*CDIM);
  for (int ti = wv; ti < 24; ti += 8) {
    const int mt = ti / 6, ot = __builtin_amdgcn_readfirstlane(ti % 6);
    const int mb = mt * 16, ob = ot * 16;
    v8f c = splat8(proj_b[ob + (lane & 15)]);
    #pragma unroll
    for (int kb = 0; kb < CDIM; kb += 32) {
      v16h a = frag_a(Oh,    CDIM, mb, kb, lane);
      v16h b = frag_b(projw, CDIM, ob, kb, lane);  // B[k][o] = Wproj[o][k]
      c = wmma16(a, b, c);
    }
    const int oc = ob + (lane & 15);
    const int t0 = mb + ((lane >> 4) << 3);
    #pragma unroll
    for (int r = 0; r < 8; ++r)
      og[(t0 + r) * CDIM + oc] = c[r];
  }
}

extern "C" void kernel_launch(void* const* d_in, const int* in_sizes, int n_in,
                              void* d_out, int out_size, void* d_ws, size_t ws_size,
                              hipStream_t stream) {
  const float* x          = (const float*)d_in[0];
  const float* mask       = (const float*)d_in[1];
  const float* qkv_w      = (const float*)d_in[2];
  const float* qkv_b      = (const float*)d_in[3];
  const float* proj_w     = (const float*)d_in[4];
  const float* proj_b     = (const float*)d_in[5];
  const float* bias_table = (const float*)d_in[6];
  const int*   rel_idx    = (const int*)d_in[7];
  float* out = (float*)d_out;

  // Workspace carve: f16 qkv_w | f16 proj_w | f32 bias3  (~120 KB total)
  char* ws = (char*)d_ws;
  _Float16* qkvw_h  = (_Float16*)ws;
  _Float16* projw_h = (_Float16*)(ws + (size_t)QKV_W_ELEMS * 2);
  float*    bias3   = (float*)(ws + (size_t)QKV_W_ELEMS * 2 + (size_t)PROJ_W_ELEMS * 2);

  const int nwin  = in_sizes[0] / (NTOK * CDIM);   // 8192
  const int nmask = in_sizes[1] / (NTOK * NTOK);   // 4096

  const int prep_total = QKV_W_ELEMS + PROJ_W_ELEMS + BIAS3_ELEMS;
  winattn_prep_kernel<<<(prep_total + 255) / 256, 256, 0, stream>>>(
      qkv_w, proj_w, bias_table, rel_idx, qkvw_h, projw_h, bias3);

  // Dynamic LDS: Xh+Qh+Kh+Vt (4*12288B) + S (49152B) + P (24576B) + Oh (12288B)
  const size_t smem = (size_t)4 * NTOK * CDIM * 2
                    + (size_t)NHEAD * NTOK * NTOK * 4
                    + (size_t)NHEAD * NTOK * NTOK * 2
                    + (size_t)NTOK * CDIM * 2;     // 135168 B (< 320 KB/WGP)

  winattn_kernel<<<nwin, 256, smem, stream>>>(
      x, mask, qkv_b, proj_b, qkvw_h, projw_h, bias3, out, nmask);
}
